// LabelCrossAttention_84911503442470
// MI455X (gfx1250) — matrix operand order repaired and checked by reading, not compiled, not run on hardware
//
#include <hip/hip_runtime.h>
#include <hip/hip_bf16.h>

typedef __attribute__((ext_vector_type(16))) __bf16 v16bf;
typedef __attribute__((ext_vector_type(8)))  float  v8f;

#define BB 4
#define SS 4096
#define DD 768
#define CC 8000

union FragBF {
    v16bf v;
    __bf16 b[16];
    unsigned short u[16];
    uint4 q[2];
};

__device__ __forceinline__ unsigned short bfbits(float f) {
    __bf16 h = (__bf16)f;                      // native v_cvt on gfx1250
    return __builtin_bit_cast(unsigned short, h);
}

// ---------------------------------------------------------------------------
// Kernel 1: K = x*Wk^T, V = x*Wv^T + bv  (bf16 outputs; V stored transposed)
// Grid: (M/16 = 1024, 6), block 256 (8 waves). Wave w -> e-tile blockIdx.y*8+w
// ---------------------------------------------------------------------------
__global__ __launch_bounds__(256) void kv_gemm(
    const float* __restrict__ x, const float* __restrict__ Wk,
    const float* __restrict__ Wv, const float* __restrict__ bv,
    unsigned short* __restrict__ Kbf, unsigned short* __restrict__ Vt)
{
    const int tid  = threadIdx.x;
    const int wave = tid >> 5;
    const int lane = tid & 31;
    const int half = lane >> 4;
    const int ln   = lane & 15;

    const int m_base = blockIdx.x * 16;
    const int e_base = (blockIdx.y * 8 + wave) * 16;

    const float* xr  = x  + (size_t)(m_base + ln) * DD;   // A rows
    const float* wkr = Wk + (size_t)(e_base + ln) * DD;   // B rows (contig in d)
    const float* wvr = Wv + (size_t)(e_base + ln) * DD;

    v8f accK = {};
    v8f accV = {};

    for (int kc = 0; kc < DD / 32; ++kc) {
        // A fragment from x (f32 -> bf16): 16 contiguous f32 along d at kc*32 + half*16
        FragBF fa;
        {
            const float* p = xr + kc * 32 + half * 16;
            float4 c0 = *(const float4*)(p + 0);
            float4 c1 = *(const float4*)(p + 4);
            float4 c2 = *(const float4*)(p + 8);
            float4 c3 = *(const float4*)(p + 12);
            fa.b[0]=(__bf16)c0.x; fa.b[1]=(__bf16)c0.y; fa.b[2]=(__bf16)c0.z; fa.b[3]=(__bf16)c0.w;
            fa.b[4]=(__bf16)c1.x; fa.b[5]=(__bf16)c1.y; fa.b[6]=(__bf16)c1.z; fa.b[7]=(__bf16)c1.w;
            fa.b[8]=(__bf16)c2.x; fa.b[9]=(__bf16)c2.y; fa.b[10]=(__bf16)c2.z; fa.b[11]=(__bf16)c2.w;
            fa.b[12]=(__bf16)c3.x; fa.b[13]=(__bf16)c3.y; fa.b[14]=(__bf16)c3.z; fa.b[15]=(__bf16)c3.w;
        }
        // B fragments from Wk / Wv: 16 contiguous f32 along d at kc*32 + half*16
        FragBF fbk, fbv;
        {
            const float* p = wkr + kc * 32 + half * 16;
            float4 c0 = *(const float4*)(p + 0);
            float4 c1 = *(const float4*)(p + 4);
            float4 c2 = *(const float4*)(p + 8);
            float4 c3 = *(const float4*)(p + 12);
            fbk.b[0]=(__bf16)c0.x; fbk.b[1]=(__bf16)c0.y; fbk.b[2]=(__bf16)c0.z; fbk.b[3]=(__bf16)c0.w;
            fbk.b[4]=(__bf16)c1.x; fbk.b[5]=(__bf16)c1.y; fbk.b[6]=(__bf16)c1.z; fbk.b[7]=(__bf16)c1.w;
            fbk.b[8]=(__bf16)c2.x; fbk.b[9]=(__bf16)c2.y; fbk.b[10]=(__bf16)c2.z; fbk.b[11]=(__bf16)c2.w;
            fbk.b[12]=(__bf16)c3.x; fbk.b[13]=(__bf16)c3.y; fbk.b[14]=(__bf16)c3.z; fbk.b[15]=(__bf16)c3.w;
        }
        {
            const float* p = wvr + kc * 32 + half * 16;
            float4 c0 = *(const float4*)(p + 0);
            float4 c1 = *(const float4*)(p + 4);
            float4 c2 = *(const float4*)(p + 8);
            float4 c3 = *(const float4*)(p + 12);
            fbv.b[0]=(__bf16)c0.x; fbv.b[1]=(__bf16)c0.y; fbv.b[2]=(__bf16)c0.z; fbv.b[3]=(__bf16)c0.w;
            fbv.b[4]=(__bf16)c1.x; fbv.b[5]=(__bf16)c1.y; fbv.b[6]=(__bf16)c1.z; fbv.b[7]=(__bf16)c1.w;
            fbv.b[8]=(__bf16)c2.x; fbv.b[9]=(__bf16)c2.y; fbv.b[10]=(__bf16)c2.z; fbv.b[11]=(__bf16)c2.w;
            fbv.b[12]=(__bf16)c3.x; fbv.b[13]=(__bf16)c3.y; fbv.b[14]=(__bf16)c3.z; fbv.b[15]=(__bf16)c3.w;
        }
        accK = __builtin_amdgcn_wmma_f32_16x16x32_bf16(false, fa.v, false, fbk.v,
                                                       (short)0, accK, false, false);
        accV = __builtin_amdgcn_wmma_f32_16x16x32_bf16(false, fa.v, false, fbv.v,
                                                       (short)0, accV, false, false);
    }

    const int col = e_base + ln;
    const float bvc = bv[col];
#pragma unroll
    for (int i = 0; i < 8; ++i) {
        const int row = m_base + i + half * 8;       // global bs index
        const int b = row >> 12;                     // / 4096
        const int s = row & (SS - 1);
        Kbf[(size_t)row * DD + col] = bfbits(accK[i]);
        Vt[(((size_t)b * DD + col) << 12) + s] = bfbits(accV[i] + bvc);
    }
}

// ---------------------------------------------------------------------------
// Kernel 2: label_reps f32 -> bf16
// ---------------------------------------------------------------------------
__global__ __launch_bounds__(256) void q_convert(const float* __restrict__ src,
                                                 unsigned short* __restrict__ dst)
{
    const size_t idx = ((size_t)blockIdx.x * 256 + threadIdx.x) * 4;
    if (idx >= (size_t)CC * DD) return;
    float4 f = *(const float4*)(src + idx);
    uint2 p;
    p.x = (unsigned int)bfbits(f.x) | ((unsigned int)bfbits(f.y) << 16);
    p.y = (unsigned int)bfbits(f.z) | ((unsigned int)bfbits(f.w) << 16);
    *(uint2*)(dst + idx) = p;
}

// ---------------------------------------------------------------------------
// Kernel 3: fused flash attention + LayerNorm-reduction + output dot.
// Grid: (C/16 = 500, B = 4), block 256 = 8 waves.
// Wave w: scores for s-subtile [s0+w*16, +16); accumulator slice d in [w*96,+96).
// ---------------------------------------------------------------------------
__global__ __launch_bounds__(256) void attn_fused(
    const unsigned short* __restrict__ Kbf, const unsigned short* __restrict__ Vt,
    const unsigned short* __restrict__ Qbf, const int* __restrict__ amask,
    const float* __restrict__ ln_gamma, const float* __restrict__ ln_beta,
    const float* __restrict__ out_w, const float* __restrict__ out_b,
    float* __restrict__ dout)
{
    __shared__ unsigned short Qlds[16 * DD];     // 24 KB: Q tile, bf16
    __shared__ unsigned short Plds[16 * 128];    // 4 KB: P tile, bf16
    __shared__ float redMax[8][16];
    __shared__ float redSum[8][16];
    __shared__ float mrow[16], lrow[16], arow[16];
    __shared__ float s1[16], s2[16], s3[16];
    __shared__ float wgbS, wbS;

    const int tid  = threadIdx.x;
    const int wave = tid >> 5;
    const int lane = tid & 31;
    const int half = lane >> 4;
    const int ln   = lane & 15;

    const int b      = blockIdx.y;
    const int c_base = blockIdx.x * 16;

    // ---- load Q tile to LDS (bf16, row-major [16][768]) ----
    {
        const unsigned int* src = (const unsigned int*)(Qbf + (size_t)c_base * DD);
        unsigned int* dst = (unsigned int*)Qlds;
        for (int j = tid; j < 16 * DD / 2; j += 256) dst[j] = src[j];
    }
    if (tid < 16) {
        mrow[tid] = -1e30f; lrow[tid] = 0.0f;
        s1[tid] = 0.0f; s2[tid] = 0.0f; s3[tid] = 0.0f;
    }
    if (tid == 0) { wgbS = 0.0f; wbS = 0.0f; }
    __syncthreads();
    {   // constants Σ w*γ and Σ w*β (once per block)
        float pg = 0.0f, pb = 0.0f;
        for (int d = tid; d < DD; d += 256) {
            float w = out_w[d];
            pg += w * ln_gamma[d];
            pb += w * ln_beta[d];
        }
        atomicAdd(&wgbS, pg);
        atomicAdd(&wbS, pb);
    }

    v8f acc[6];
    const v8f vzero = {};
#pragma unroll
    for (int t = 0; t < 6; ++t) acc[t] = vzero;

    for (int s0 = 0; s0 < SS; s0 += 128) {
        // ---- scores: 16c x 16s tile per wave (K-dim = D, 24 WMMA) ----
        v8f sc = vzero;
        const int scol = s0 + wave * 16 + ln;
        const unsigned short* krow =
            Kbf + ((size_t)(b * SS + scol)) * DD + half * 16;
        // prefetch next S-tile K/V rows into L2 (global_prefetch_b8)
        if (s0 + 128 < SS) {
            __builtin_prefetch(krow + (size_t)128 * DD, 0, 1);
            __builtin_prefetch(Vt + (((size_t)b * DD + (wave * 96 + ln)) << 12)
                                  + s0 + 128, 0, 1);
        }
#pragma unroll 4
        for (int kc = 0; kc < DD / 32; ++kc) {
            FragBF a, bk;
            const unsigned short* qp = Qlds + ln * DD + kc * 32 + half * 16;
            a.q[0] = *(const uint4*)qp;
            a.q[1] = *(const uint4*)(qp + 8);
            const unsigned short* kp = krow + kc * 32;
            bk.q[0] = *(const uint4*)kp;
            bk.q[1] = *(const uint4*)(kp + 8);
            sc = __builtin_amdgcn_wmma_f32_16x16x32_bf16(false, a.v, false, bk.v,
                                                         (short)0, sc, false, false);
        }
        // ---- mask (column scol shared by all 8 rows of this lane) ----
        const int mv = amask[b * SS + scol];
        float scf[8];
#pragma unroll
        for (int i = 0; i < 8; ++i) scf[i] = mv ? sc[i] : -1e30f;

        // ---- row max: butterfly within 16-lane half, then cross-wave ----
        float rm[8];
#pragma unroll
        for (int i = 0; i < 8; ++i) {
            float m = scf[i];
            m = fmaxf(m, __shfl_xor(m, 1, 32));
            m = fmaxf(m, __shfl_xor(m, 2, 32));
            m = fmaxf(m, __shfl_xor(m, 4, 32));
            m = fmaxf(m, __shfl_xor(m, 8, 32));
            rm[i] = m;
        }
        if (ln == 0) {
#pragma unroll
            for (int i = 0; i < 8; ++i) redMax[wave][half * 8 + i] = rm[i];
        }
        __syncthreads();
        if (tid < 16) {
            float mo = mrow[tid];
            float mn = mo;
#pragma unroll
            for (int w2 = 0; w2 < 8; ++w2) mn = fmaxf(mn, redMax[w2][tid]);
            arow[tid] = __expf(mo - mn);
            mrow[tid] = mn;
        }
        __syncthreads();

        // ---- p = exp(s - m_new); tile row sums; write P; rescale acc ----
        float mn_i[8], al_i[8];
#pragma unroll
        for (int i = 0; i < 8; ++i) {
            mn_i[i] = mrow[half * 8 + i];
            al_i[i] = arow[half * 8 + i];
        }
        float ps[8], rs[8];
#pragma unroll
        for (int i = 0; i < 8; ++i) ps[i] = __expf(scf[i] - mn_i[i]);
#pragma unroll
        for (int i = 0; i < 8; ++i) {
            float s = ps[i];
            s += __shfl_xor(s, 1, 32);
            s += __shfl_xor(s, 2, 32);
            s += __shfl_xor(s, 4, 32);
            s += __shfl_xor(s, 8, 32);
            rs[i] = s;
        }
        if (ln == 0) {
#pragma unroll
            for (int i = 0; i < 8; ++i) redSum[wave][half * 8 + i] = rs[i];
        }
#pragma unroll
        for (int i = 0; i < 8; ++i)
            Plds[(i + half * 8) * 128 + wave * 16 + ln] = bfbits(ps[i]);
#pragma unroll
        for (int t = 0; t < 6; ++t)
#pragma unroll
            for (int i = 0; i < 8; ++i) acc[t][i] *= al_i[i];
        __syncthreads();
        if (tid < 16) {
            float l = lrow[tid] * arow[tid];
#pragma unroll
            for (int w2 = 0; w2 < 8; ++w2) l += redSum[w2][tid];
            lrow[tid] = l;
        }

        // ---- acc += P (16c x 128s) * V (128s x 96d-slice): 24 WMMA ----
#pragma unroll
        for (int t = 0; t < 6; ++t) {
            const unsigned short* vrow =
                Vt + (((size_t)b * DD + (wave * 96 + t * 16 + ln)) << 12)
                   + s0 + half * 16;
#pragma unroll
            for (int kc = 0; kc < 4; ++kc) {
                FragBF pa, vb;
                const unsigned short* pp = Plds + ln * 128 + kc * 32 + half * 16;
                pa.q[0] = *(const uint4*)pp;
                pa.q[1] = *(const uint4*)(pp + 8);
                const unsigned short* vp = vrow + kc * 32;
                vb.q[0] = *(const uint4*)vp;
                vb.q[1] = *(const uint4*)(vp + 8);
                acc[t] = __builtin_amdgcn_wmma_f32_16x16x32_bf16(false, pa.v, false, vb.v,
                                                                 (short)0, acc[t], false, false);
            }
        }
        __syncthreads();
    }

    // ---- finalize: y = acc/l; reduce Σy, Σy², Σ(wγ)y per class row ----
    float linv[8];
#pragma unroll
    for (int i = 0; i < 8; ++i) linv[i] = 1.0f / lrow[half * 8 + i];

    float p1[8], p2[8], p3[8];
#pragma unroll
    for (int i = 0; i < 8; ++i) { p1[i] = 0.0f; p2[i] = 0.0f; p3[i] = 0.0f; }
#pragma unroll
    for (int t = 0; t < 6; ++t) {
        const int d = wave * 96 + t * 16 + ln;
        const float wg = out_w[d] * ln_gamma[d];
#pragma unroll
        for (int i = 0; i < 8; ++i) {
            float y = acc[t][i] * linv[i];
            p1[i] += y;
            p2[i] += y * y;
            p3[i] += wg * y;
        }
    }
#pragma unroll
    for (int i = 0; i < 8; ++i) {
        float a1 = p1[i], a2 = p2[i], a3 = p3[i];
        a1 += __shfl_xor(a1, 1, 32); a2 += __shfl_xor(a2, 1, 32); a3 += __shfl_xor(a3, 1, 32);
        a1 += __shfl_xor(a1, 2, 32); a2 += __shfl_xor(a2, 2, 32); a3 += __shfl_xor(a3, 2, 32);
        a1 += __shfl_xor(a1, 4, 32); a2 += __shfl_xor(a2, 4, 32); a3 += __shfl_xor(a3, 4, 32);
        a1 += __shfl_xor(a1, 8, 32); a2 += __shfl_xor(a2, 8, 32); a3 += __shfl_xor(a3, 8, 32);
        p1[i] = a1; p2[i] = a2; p3[i] = a3;
    }
    if (ln == 0) {
#pragma unroll
        for (int i = 0; i < 8; ++i) {
            const int r = half * 8 + i;
            atomicAdd(&s1[r], p1[i]);
            atomicAdd(&s2[r], p2[i]);
            atomicAdd(&s3[r], p3[i]);
        }
    }
    __syncthreads();
    if (tid < 16) {
        const float invD = 1.0f / (float)DD;
        float mu  = s1[tid] * invD;
        float var = s2[tid] * invD - mu * mu;
        float rin = rsqrtf(var + 1e-5f);
        float o = rin * (s3[tid] - mu * wgbS) + wbS + out_b[0];
        dout[b * CC + c_base + tid] = o;
    }
}

// ---------------------------------------------------------------------------
extern "C" void kernel_launch(void* const* d_in, const int* in_sizes, int n_in,
                              void* d_out, int out_size, void* d_ws, size_t ws_size,
                              hipStream_t stream)
{
    const float* x     = (const float*)d_in[0];
    const int*   amask = (const int*)d_in[1];
    const float* Wk    = (const float*)d_in[2];
    const float* Wv    = (const float*)d_in[3];
    const float* bv    = (const float*)d_in[4];
    const float* lrep  = (const float*)d_in[5];
    const float* lng   = (const float*)d_in[6];
    const float* lnb   = (const float*)d_in[7];
    const float* outw  = (const float*)d_in[8];
    const float* outb  = (const float*)d_in[9];
    float* dout = (float*)d_out;

    unsigned short* ws  = (unsigned short*)d_ws;
    unsigned short* Kbf = ws;                                  // [B,S,D] bf16
    unsigned short* Vt  = Kbf + (size_t)BB * SS * DD;          // [B,D,S] bf16
    unsigned short* Qbf = Vt  + (size_t)BB * SS * DD;          // [C,D]   bf16

    kv_gemm<<<dim3((BB * SS) / 16, DD / (16 * 8)), 256, 0, stream>>>(
        x, Wk, Wv, bv, Kbf, Vt);
    q_convert<<<dim3((unsigned)(((size_t)CC * DD / 4 + 255) / 256)), 256, 0, stream>>>(lrep, Qbf);
    attn_fused<<<dim3(CC / 16, BB), 256, 0, stream>>>(
        Kbf, Vt, Qbf, amask, lng, lnb, outw, outb, dout);
}
